// TransformerBlock_26079041421337
// MI455X (gfx1250) — matrix operand order repaired and checked by reading, not compiled, not run on hardware
//
#include <hip/hip_runtime.h>
#include <hip/hip_bf16.h>
#include <cmath>

// Problem dims
#define BB  2
#define SS  2048
#define DDim 1024
#define HH  16
#define FFD 4096
#define DKK 64

typedef __attribute__((ext_vector_type(16))) __bf16 v16bf;
typedef __attribute__((ext_vector_type(8)))  float  v8f;
typedef __attribute__((ext_vector_type(4)))  float  v4f;

union Pack { v4f f[2]; v16bf v; };

// A-operand (16x32 bf16): lane holds row m = lane&15, kbase = 2*kb = (lane>>4)*16.
// VGPR0-7 = K [2kb..2kb+15]  (16 contiguous K values, matching load_b32)
__device__ inline v16bf load_a32(const __bf16* row, int kb) {
  Pack u;
  u.f[0] = *reinterpret_cast<const v4f*>(row + 2 * kb);
  u.f[1] = *reinterpret_cast<const v4f*>(row + 2 * kb + 8);
  return u.v;
}
// B-operand (32x16 bf16): lane holds col n = lane&15, kbase = (lane>>4)*16.
// 16 contiguous K values (32B) from the (N,K)-row-major matrix's row n.
__device__ inline v16bf load_b32(const __bf16* row, int kb) {
  Pack u;
  u.f[0] = *reinterpret_cast<const v4f*>(row + kb);
  u.f[1] = *reinterpret_cast<const v4f*>(row + kb + 8);
  return u.v;
}

// Async 16B global -> LDS copy (CDNA5 GLOBAL_LOAD_ASYNC_TO_LDS_B128, GVS mode:
// mem = SADDR(64b sgpr) + VADDR(32b voffset); LDS dst = VGPR lds byte address).
// Tracked by ASYNCcnt.
__device__ inline void async_copy16(unsigned int lds_addr, const void* sbase,
                                    unsigned int voff) {
  asm volatile("global_load_async_to_lds_b128 %0, %1, %2"
               :: "v"(lds_addr), "v"(voff), "s"(sbase)
               : "memory");
}

// ---------------------------------------------------------------------------
// fp32 -> bf16 conversion
// ---------------------------------------------------------------------------
__global__ __launch_bounds__(256) void cvt_bf16(const float* __restrict__ in,
                                                __bf16* __restrict__ out, int n) {
  int i = blockIdx.x * 256 + threadIdx.x;
  if (i < n) out[i] = (__bf16)in[i];
}

// ---------------------------------------------------------------------------
// RMSNorm: one 256-thread block per row of D=1024, output bf16
// ---------------------------------------------------------------------------
__global__ __launch_bounds__(256) void rmsnorm_k(const float* __restrict__ x,
                                                 const float* __restrict__ g,
                                                 __bf16* __restrict__ out) {
  __shared__ float red[256];
  const int row = blockIdx.x;
  const float* xr = x + (size_t)row * DDim;
  float s = 0.f;
  for (int i = threadIdx.x; i < DDim; i += 256) { float v = xr[i]; s += v * v; }
  red[threadIdx.x] = s;
  __syncthreads();
  for (int off = 128; off > 0; off >>= 1) {
    if ((int)threadIdx.x < off) red[threadIdx.x] += red[threadIdx.x + off];
    __syncthreads();
  }
  float inv = rsqrtf(red[0] / (float)DDim + 1e-5f);
  __bf16* orow = out + (size_t)row * DDim;
  for (int i = threadIdx.x; i < DDim; i += 256)
    orow[i] = (__bf16)(xr[i] * inv * g[i]);
}

// ---------------------------------------------------------------------------
// NT GEMM: C[M,N] = A[M,K] * W[N,K]^T  (A,W bf16 row-major, acc f32)
// 128x128 block tile, K-step 32. A/B tiles double-buffered in LDS via
// global_load_async_to_lds_b128 (ASYNCcnt). 8 waves, wave tile 32x64.
// Epilogue: optional silu-gate combine (aux), optional residual, f32/bf16 out.
// ---------------------------------------------------------------------------
__global__ __launch_bounds__(256) void gemm_nt_bf16(
    const __bf16* __restrict__ A, const __bf16* __restrict__ W,
    const float* __restrict__ residual, const float* __restrict__ silu_aux,
    float* __restrict__ outF, __bf16* __restrict__ outBF,
    int M, int N, int K) {
  // 2 buffers x 128 rows x 32 bf16 (64B rows) for each of A and B
  __shared__ __attribute__((aligned(16))) __bf16 Abuf[2][128 * 32];
  __shared__ __attribute__((aligned(16))) __bf16 Bbuf[2][128 * 32];

  const int tid  = threadIdx.x;
  const int lane = tid & 31;
  const int wave = tid >> 5;
  const int lh   = lane >> 4;       // lane half
  const int l15  = lane & 15;
  const int m0blk = blockIdx.y * 128;
  const int n0blk = blockIdx.x * 128;
  const int mrow = (wave >> 1) * 32;   // wave tile origin inside block tile
  const int nrow = (wave & 1) * 64;
  (void)M;

  // stage a 128x32 tile of A and of W at column kk into LDS buffer `buf`
  auto stage = [&](int buf, int kk) {
    const __bf16* Ab = A + (size_t)m0blk * K + kk;
    const __bf16* Bb = W + (size_t)n0blk * K + kk;
#pragma unroll
    for (int i = 0; i < 2; ++i) {
      int c   = tid + i * 256;          // 0..511 : 16B chunk id
      int row = c >> 2;                 // 0..127
      int q   = c & 3;                  // 16B chunk within 64B row
      unsigned int goff = (unsigned int)((row * K + q * 8) * 2);  // bytes
      unsigned int la = (unsigned int)(uintptr_t)&Abuf[buf][0] + row * 64 + q * 16;
      unsigned int lb = (unsigned int)(uintptr_t)&Bbuf[buf][0] + row * 64 + q * 16;
      async_copy16(la, (const void*)Ab, goff);
      async_copy16(lb, (const void*)Bb, goff);
    }
  };

  v8f acc[2][4] = {};
  const int nk = K / 32;

  stage(0, 0);
  for (int ki = 0; ki < nk; ++ki) {
    const int buf = ki & 1;
    if (ki + 1 < nk) {
      stage(buf ^ 1, (ki + 1) * 32);
      asm volatile("s_wait_asynccnt 0x4" ::: "memory"); // stage ki complete
    } else {
      asm volatile("s_wait_asynccnt 0x0" ::: "memory");
    }
    __syncthreads();

    v16bf a[2], b[4];
#pragma unroll
    for (int ms = 0; ms < 2; ++ms)
      a[ms] = load_a32(&Abuf[buf][(mrow + ms * 16 + l15) * 32], lh * 8);
#pragma unroll
    for (int ns = 0; ns < 4; ++ns)
      b[ns] = load_b32(&Bbuf[buf][(nrow + ns * 16 + l15) * 32], lh * 16);
#pragma unroll
    for (int ms = 0; ms < 2; ++ms)
#pragma unroll
      for (int ns = 0; ns < 4; ++ns)
        acc[ms][ns] = __builtin_amdgcn_wmma_f32_16x16x32_bf16(
            false, a[ms], false, b[ns], (short)0, acc[ms][ns], false, false);

    __syncthreads();   // all waves done reading before buffer is re-staged
  }

  // C/D layout: vgpr r <-> row (r + 8*lanehalf), col = lane&15
#pragma unroll
  for (int ms = 0; ms < 2; ++ms)
#pragma unroll
    for (int ns = 0; ns < 4; ++ns)
#pragma unroll
      for (int r = 0; r < 8; ++r) {
        int m = m0blk + mrow + ms * 16 + r + lh * 8;
        int n = n0blk + nrow + ns * 16 + l15;
        size_t idx = (size_t)m * N + n;
        float v = acc[ms][ns][r];
        if (silu_aux) {
          float aa = silu_aux[idx];
          v *= aa * __builtin_amdgcn_rcpf(1.0f + __expf(-aa));  // silu(aa)
        }
        if (residual) v += residual[idx];
        if (outF)  outF[idx]  = v;
        if (outBF) outBF[idx] = (__bf16)v;
      }
}

// ---------------------------------------------------------------------------
// RoPE + layout change: Qf/Kf/Vf fp32 [tok, D] ->
//   Qb/Kb bf16 [B,H,S,DK], Vt bf16 [B,H,DK,S]
// One thread per (bh, s, pair i) -- DK/2 = 32 pairs.
// ---------------------------------------------------------------------------
__global__ __launch_bounds__(256) void rope_k(const float* __restrict__ Qf,
    const float* __restrict__ Kf, const float* __restrict__ Vf,
    const int* __restrict__ tp,
    __bf16* __restrict__ Qb, __bf16* __restrict__ Kb, __bf16* __restrict__ Vt) {
  int idx = blockIdx.x * 256 + threadIdx.x;   // over B*H*S*32
  int i   = idx & 31;
  int row = idx >> 5;            // bh*S + s
  int sq  = row % SS;
  int bh  = row / SS;
  int b   = bh / HH, h = bh % HH;
  int tok = b * SS + sq;
  float p = (float)tp[tok];
  // inv_freq_i = theta^(-2i/DK) = exp(-(i/32)*ln(10000))
  float freq = p * __expf(-0.28782313662425572f * (float)i);
  float sn, cs;
  __sincosf(freq, &sn, &cs);
  size_t gi = (size_t)tok * DDim + (size_t)h * DKK + 2 * i;
  float qr = Qf[gi], qi = Qf[gi + 1];
  float kr = Kf[gi], ki = Kf[gi + 1];
  float vr = Vf[gi], vi = Vf[gi + 1];
  size_t oi = ((size_t)bh * SS + sq) * DKK + 2 * i;
  Qb[oi]     = (__bf16)(qr * cs - qi * sn);
  Qb[oi + 1] = (__bf16)(qr * sn + qi * cs);
  Kb[oi]     = (__bf16)(kr * cs - ki * sn);
  Kb[oi + 1] = (__bf16)(kr * sn + ki * cs);
  size_t vo = ((size_t)bh * DKK + 2 * i) * SS + sq;
  Vt[vo]      = (__bf16)vr;
  Vt[vo + SS] = (__bf16)vi;
}

// ---------------------------------------------------------------------------
// Causal flash attention. 1 wave (32 threads) per (b,h,16-row Q tile).
// kv blocks of 32: 4 WMMAs for S=Q*K^T, 4 WMMAs for P*V, online softmax.
// ---------------------------------------------------------------------------
__global__ __launch_bounds__(32) void attn_fa(const __bf16* __restrict__ Q,
    const __bf16* __restrict__ Kmat, const __bf16* __restrict__ Vt,
    __bf16* __restrict__ attn) {
  __shared__ __attribute__((aligned(16))) __bf16 Pl[16 * 32];
  const int lane = threadIdx.x;
  const int lh   = lane >> 4;
  const int l15  = lane & 15;
  const int nqt  = SS / 16;
  const int qt   = blockIdx.x % nqt;
  const int bh   = blockIdx.x / nqt;
  const int b    = bh / HH, h = bh % HH;
  const int qbase = qt * 16;
  const __bf16* Qp = Q    + (size_t)bh * SS * DKK;
  const __bf16* Kp = Kmat + (size_t)bh * SS * DKK;
  const __bf16* Vp = Vt   + (size_t)bh * DKK * SS;

  // Q A-operands for DK chunks [0,32) and [32,64)
  v16bf qa[2];
  qa[0] = load_a32(Qp + (size_t)(qbase + l15) * DKK, lh * 8);
  qa[1] = load_a32(Qp + (size_t)(qbase + l15) * DKK + 32, lh * 8);

  v8f o[4] = {};
  float rm[8], rl[8];
#pragma unroll
  for (int r = 0; r < 8; ++r) { rm[r] = -INFINITY; rl[r] = 0.0f; }

  for (int kv = 0; kv < qbase + 16; kv += 32) {
    v8f s0 = {}, s1 = {};
#pragma unroll
    for (int c = 0; c < 2; ++c) {
      v16bf k0 = load_b32(Kp + (size_t)(kv + l15) * DKK + c * 32, lh * 16);
      v16bf k1 = load_b32(Kp + (size_t)(kv + 16 + l15) * DKK + c * 32, lh * 16);
      s0 = __builtin_amdgcn_wmma_f32_16x16x32_bf16(false, qa[c], false, k0,
                                                   (short)0, s0, false, false);
      s1 = __builtin_amdgcn_wmma_f32_16x16x32_bf16(false, qa[c], false, k1,
                                                   (short)0, s1, false, false);
    }
#pragma unroll
    for (int r = 0; r < 8; ++r) {
      int qrow = qbase + r + lh * 8;
      float v0 = s0[r] * 0.125f;     // 1/sqrt(DK)
      float v1 = s1[r] * 0.125f;
      if (kv + l15 > qrow)      v0 = -1e30f;   // causal mask
      if (kv + 16 + l15 > qrow) v1 = -1e30f;

      float mx = fmaxf(v0, v1);
      for (int off = 8; off > 0; off >>= 1) mx = fmaxf(mx, __shfl_xor(mx, off, 16));
      float mnew = fmaxf(rm[r], mx);
      float alpha = __expf(rm[r] - mnew);
      float p0 = __expf(v0 - mnew);
      float p1 = __expf(v1 - mnew);
      float ps = p0 + p1;
      for (int off = 8; off > 0; off >>= 1) ps += __shfl_xor(ps, off, 16);
      rl[r] = rl[r] * alpha + ps;
      rm[r] = mnew;
#pragma unroll
      for (int ns = 0; ns < 4; ++ns) o[ns][r] *= alpha;
      Pl[(r + lh * 8) * 32 + l15]      = (__bf16)p0;
      Pl[(r + lh * 8) * 32 + 16 + l15] = (__bf16)p1;
    }
    asm volatile("s_wait_dscnt 0" ::: "memory");
    // re-read P (16x32) in A-operand layout from LDS
    v16bf pa = load_a32((const __bf16*)Pl + l15 * 32, lh * 8);
#pragma unroll
    for (int ns = 0; ns < 4; ++ns) {
      v16bf vb = load_b32(Vp + (size_t)(ns * 16 + l15) * SS + kv, lh * 16);
      o[ns] = __builtin_amdgcn_wmma_f32_16x16x32_bf16(false, pa, false, vb,
                                                      (short)0, o[ns], false, false);
    }
  }

#pragma unroll
  for (int r = 0; r < 8; ++r) {
    float rinv = __builtin_amdgcn_rcpf(rl[r]);
    int row  = qbase + r + lh * 8;
    int tok  = b * SS + row;
#pragma unroll
    for (int ns = 0; ns < 4; ++ns) {
      int feat = h * DKK + ns * 16 + l15;
      attn[(size_t)tok * DDim + feat] = (__bf16)(o[ns][r] * rinv);
    }
  }
}

// ---------------------------------------------------------------------------
extern "C" void kernel_launch(void* const* d_in, const int* in_sizes, int n_in,
                              void* d_out, int out_size, void* d_ws, size_t ws_size,
                              hipStream_t stream) {
  (void)in_sizes; (void)n_in; (void)out_size; (void)ws_size;
  const float* x  = (const float*)d_in[0];
  const float* wq = (const float*)d_in[1];
  const float* wk = (const float*)d_in[2];
  const float* wv = (const float*)d_in[3];
  const float* wo = (const float*)d_in[4];
  const float* w1 = (const float*)d_in[5];
  const float* w2 = (const float*)d_in[6];
  const float* w3 = (const float*)d_in[7];
  const float* g1 = (const float*)d_in[8];
  const float* g2 = (const float*)d_in[9];
  const int*   tp = (const int*)d_in[10];

  const int M = BB * SS;                      // 4096 tokens
  char* base = (char*)d_ws;
  size_t off = 0;
  auto alloc = [&](size_t bytes) -> void* {
    void* p = base + off;
    off += (bytes + 255) & ~(size_t)255;
    return p;
  };

  __bf16* wq_b = (__bf16*)alloc((size_t)DDim * DDim * 2);
  __bf16* wk_b = (__bf16*)alloc((size_t)DDim * DDim * 2);
  __bf16* wv_b = (__bf16*)alloc((size_t)DDim * DDim * 2);
  __bf16* wo_b = (__bf16*)alloc((size_t)DDim * DDim * 2);
  __bf16* w1_b = (__bf16*)alloc((size_t)FFD * DDim * 2);
  __bf16* w3_b = (__bf16*)alloc((size_t)FFD * DDim * 2);
  __bf16* w2_b = (__bf16*)alloc((size_t)DDim * FFD * 2);
  __bf16* h_b  = (__bf16*)alloc((size_t)M * DDim * 2);      // reused for h2
  __bf16* Qb   = (__bf16*)alloc((size_t)BB * HH * SS * DKK * 2);
  __bf16* Kb   = (__bf16*)alloc((size_t)BB * HH * SS * DKK * 2);
  __bf16* Vt   = (__bf16*)alloc((size_t)BB * HH * DKK * SS * 2);
  __bf16* attn_b = (__bf16*)alloc((size_t)M * DDim * 2);
  float*  yf   = (float*)alloc((size_t)M * DDim * 4);
  // union region: {Qf,Kf,Vf} (48MB, dead after rope) aliases af (64MB)
  char*   un   = (char*)alloc((size_t)M * FFD * 4);
  float*  Qf   = (float*)un;
  float*  Kf   = Qf + (size_t)M * DDim;
  float*  Vf   = Kf + (size_t)M * DDim;
  float*  af   = (float*)un;
  __bf16* ffin = (__bf16*)alloc((size_t)M * FFD * 2);

  auto cvt = [&](const float* src, __bf16* dst, size_t n) {
    cvt_bf16<<<dim3((unsigned)((n + 255) / 256)), dim3(256), 0, stream>>>(src, dst, (int)n);
  };
  cvt(wq, wq_b, (size_t)DDim * DDim);
  cvt(wk, wk_b, (size_t)DDim * DDim);
  cvt(wv, wv_b, (size_t)DDim * DDim);
  cvt(wo, wo_b, (size_t)DDim * DDim);
  cvt(w1, w1_b, (size_t)FFD * DDim);
  cvt(w3, w3_b, (size_t)FFD * DDim);
  cvt(w2, w2_b, (size_t)DDim * FFD);

  // h = rmsnorm(x, g1)
  rmsnorm_k<<<M, 256, 0, stream>>>(x, g1, h_b);

  // Q/K/V projections (fp32 out)
  gemm_nt_bf16<<<dim3(DDim / 128, M / 128), 256, 0, stream>>>(
      h_b, wq_b, nullptr, nullptr, Qf, nullptr, M, DDim, DDim);
  gemm_nt_bf16<<<dim3(DDim / 128, M / 128), 256, 0, stream>>>(
      h_b, wk_b, nullptr, nullptr, Kf, nullptr, M, DDim, DDim);
  gemm_nt_bf16<<<dim3(DDim / 128, M / 128), 256, 0, stream>>>(
      h_b, wv_b, nullptr, nullptr, Vf, nullptr, M, DDim, DDim);

  // RoPE + bf16 conversion + V transpose
  rope_k<<<(BB * HH * SS * 32) / 256, 256, 0, stream>>>(Qf, Kf, Vf, tp, Qb, Kb, Vt);

  // flash attention
  attn_fa<<<BB * HH * (SS / 16), 32, 0, stream>>>(Qb, Kb, Vt, attn_b);

  // y = x + attn @ wo^T
  gemm_nt_bf16<<<dim3(DDim / 128, M / 128), 256, 0, stream>>>(
      attn_b, wo_b, x, nullptr, yf, nullptr, M, DDim, DDim);

  // h2 = rmsnorm(y, g2)
  rmsnorm_k<<<M, 256, 0, stream>>>(yf, g2, h_b);

  // a = h2 @ w1^T (fp32)
  gemm_nt_bf16<<<dim3(FFD / 128, M / 128), 256, 0, stream>>>(
      h_b, w1_b, nullptr, nullptr, af, nullptr, M, FFD, DDim);
  // ffin = silu(a) * (h2 @ w3^T)  (bf16)
  gemm_nt_bf16<<<dim3(FFD / 128, M / 128), 256, 0, stream>>>(
      h_b, w3_b, nullptr, af, nullptr, ffin, M, FFD, DDim);
  // out = y + ffin @ w2^T
  gemm_nt_bf16<<<dim3(DDim / 128, M / 128), 256, 0, stream>>>(
      ffin, w2_b, yf, nullptr, (float*)d_out, nullptr, M, DDim, FFD);
}
